// ComGNN_64484638982550
// MI455X (gfx1250) — compile-verified, matrix operand order checked by reading, not run on hardware
//
#include <hip/hip_runtime.h>
#include <hip/hip_bf16.h>
#include <math.h>

#define N_NODES 20000
#define N_EDGES 200000
#define T_STEPS 3
#define EMB 64
#define NCONS 2

typedef __attribute__((ext_vector_type(16))) _Float16 v16h;
typedef __attribute__((ext_vector_type(4)))  _Float16 v4h;
typedef __attribute__((ext_vector_type(8)))  float    v8f;

// ---------------------------------------------------------------------------
// WMMA fragment layout (CDNA5 v_wmma_f32_16x16x32_f16):
// A 16x32 f16: lane<16 holds row M=lane, K in {kb..kb+7, kb+16..kb+23},
//   kb = 0 (lanes 0-15) / 8 (lanes 16-31); half j -> k = kb + (j&7) + ((j&8)<<1)
// B 32x16 mirrors A with lane%16 = column N.
// C/D 16x16 f32: VGPR r -> (M = r + (lane<16?0:8), N = lane%16).
//
// Weights and activation tiles live in LDS *pre-swizzled* into fragment order
// [frag][lane][16 halves], so fragment loads are contiguous 32B (2x b128).
// ---------------------------------------------------------------------------
__device__ __forceinline__ int koff(int lane, int j) {
  return ((lane & 16) ? 8 : 0) + (j & 7) + ((j & 8) << 1);
}

__device__ __forceinline__ v8f wmma16(v16h a, v16h b, v8f c) {
  return __builtin_amdgcn_wmma_f32_16x16x32_f16(false, a, false, b, (short)0, c,
                                                false, false);
}

// Branch-free activations on the hardware transcendental pipe.
__device__ __forceinline__ float fexp(float x) {
  return __builtin_amdgcn_exp2f(x * 1.4426950408889634f);
}
__device__ __forceinline__ float fast_tanh(float x) {
  return 1.0f - 2.0f * __builtin_amdgcn_rcpf(fexp(2.0f * x) + 1.0f);
}
__device__ __forceinline__ float fast_sigmoid(float x) {
  return __builtin_amdgcn_rcpf(1.0f + fexp(-x));
}

// Scatter one f32 element of a 16x64 activation tile into swizzled A-fragment
// order: tile is _Float16[2 kb][32 lane][16 half].
__device__ __forceinline__ void put_swz(_Float16* t, int m, int col, float v) {
  int kb = col >> 5;
  int kl = col & 31;
  int lhi = (kl & 8) ? 16 : 0;
  int j = (kl & 7) + ((kl & 16) ? 8 : 0);
  t[(kb * 32 + m + lhi) * 16 + j] = (_Float16)v;
}

// Packed variant: 4 consecutive columns (col4 % 4 == 0) share one fragment and
// consecutive half slots -> single 8-byte LDS store.
__device__ __forceinline__ void put_swz4(_Float16* t, int m, int col4, float4 v) {
  int kb = col4 >> 5;
  int kl = col4 & 31;
  int lhi = (kl & 8) ? 16 : 0;
  int j = (kl & 7) + ((kl & 16) ? 8 : 0);
  v4h h;
  h[0] = (_Float16)v.x;
  h[1] = (_Float16)v.y;
  h[2] = (_Float16)v.z;
  h[3] = (_Float16)v.w;
  *(v4h*)&t[(kb * 32 + m + lhi) * 16 + j] = h;
}

// Build an A fragment from a row-major global row: koff is k-consecutive for
// j=0..7 and j=8..15, so this is exactly 4 float4 loads.
__device__ __forceinline__ v16h pack16(float4 a, float4 b, float4 c, float4 d) {
  v16h v;
  v[0] = (_Float16)a.x;  v[1] = (_Float16)a.y;
  v[2] = (_Float16)a.z;  v[3] = (_Float16)a.w;
  v[4] = (_Float16)b.x;  v[5] = (_Float16)b.y;
  v[6] = (_Float16)b.z;  v[7] = (_Float16)b.w;
  v[8] = (_Float16)c.x;  v[9] = (_Float16)c.y;
  v[10] = (_Float16)c.z; v[11] = (_Float16)c.w;
  v[12] = (_Float16)d.x; v[13] = (_Float16)d.y;
  v[14] = (_Float16)d.z; v[15] = (_Float16)d.w;
  return v;
}

__device__ __forceinline__ v16h afrag_global(const float* __restrict__ row,
                                             int lane, int k0) {
  int kb = k0 + ((lane & 16) ? 8 : 0);
  const float4* p0 = (const float4*)(row + kb);
  const float4* p1 = (const float4*)(row + kb + 16);
  return pack16(p0[0], p0[1], p1[0], p1[1]);
}

__device__ __forceinline__ v16h afrag_global_diff(const float* __restrict__ ra,
                                                  const float* __restrict__ rb,
                                                  int lane, int k0) {
  int kb = k0 + ((lane & 16) ? 8 : 0);
  const float4* a0 = (const float4*)(ra + kb);
  const float4* a1 = (const float4*)(ra + kb + 16);
  const float4* b0 = (const float4*)(rb + kb);
  const float4* b1 = (const float4*)(rb + kb + 16);
  float4 x0 = a0[0], x1 = a0[1], x2 = a1[0], x3 = a1[1];
  float4 y0 = b0[0], y1 = b0[1], y2 = b1[0], y3 = b1[1];
  float4 d0 = {x0.x - y0.x, x0.y - y0.y, x0.z - y0.z, x0.w - y0.w};
  float4 d1 = {x1.x - y1.x, x1.y - y1.y, x1.z - y1.z, x1.w - y1.w};
  float4 d2 = {x2.x - y2.x, x2.y - y2.y, x2.z - y2.z, x2.w - y2.w};
  float4 d3 = {x3.x - y3.x, x3.y - y3.y, x3.z - y3.z, x3.w - y3.w};
  return pack16(d0, d1, d2, d3);
}

// Stage a 64x64 f32 weight into swizzled B-fragment order: v16h[8 frag][32 lane],
// frag = ct*2 + kb.
__device__ __forceinline__ void stage_wswz(v16h* sW, const float* __restrict__ W,
                                           int tid, int nthr) {
  _Float16* d = (_Float16*)sW;
  for (int i = tid; i < 8 * 32 * 16; i += nthr) {
    int frag = i >> 9;
    int lane = (i >> 4) & 31;
    int j = i & 15;
    int k = (frag & 1) * 32 + koff(lane, j);
    int n = (frag >> 1) * 16 + (lane & 15);
    d[i] = (_Float16)W[k * EMB + n];
  }
}

// Stage a 32x64 f32 weight (K=32): v16h[4 frag][32 lane], frag = ct.
__device__ __forceinline__ void stage_wswz32(v16h* sW, const float* __restrict__ W,
                                             int tid, int nthr) {
  _Float16* d = (_Float16*)sW;
  for (int i = tid; i < 4 * 32 * 16; i += nthr) {
    int ct = i >> 9;
    int lane = (i >> 4) & 31;
    int j = i & 15;
    int k = koff(lane, j);
    int n = ct * 16 + (lane & 15);
    d[i] = (_Float16)W[k * EMB + n];
  }
}

__device__ __forceinline__ void stage_f32(float* dst, const float* __restrict__ src,
                                          int n, int tid, int nthr) {
  for (int i = tid; i < n; i += nthr) dst[i] = src[i];
}

// 16x64 @ 64x64 GEMM: A fragments in registers, swizzled B in LDS.
__device__ __forceinline__ void gemm_frag(v16h a0, v16h a1, const v16h* sW,
                                          const float* bias, int lane, v8f acc[4]) {
#pragma unroll
  for (int ct = 0; ct < 4; ++ct) {
    float bv = bias ? bias[ct * 16 + (lane & 15)] : 0.0f;
    v8f c;
#pragma unroll
    for (int r = 0; r < 8; ++r) c[r] = bv;
    c = wmma16(a0, sW[(ct * 2 + 0) * 32 + lane], c);
    c = wmma16(a1, sW[(ct * 2 + 1) * 32 + lane], c);
    acc[ct] = c;
  }
}

// Store C-layout accumulators into a swizzled activation tile (with tanh).
// (Targets of one lane live in different rows -> unpackable b16 scatters.)
__device__ __forceinline__ void store_swz_tanh(_Float16* tile, int lane,
                                               const v8f acc[4]) {
  int col0 = lane & 15;
  int rbase = (lane & 16) ? 8 : 0;
#pragma unroll
  for (int ct = 0; ct < 4; ++ct)
#pragma unroll
    for (int r = 0; r < 8; ++r)
      put_swz(tile, rbase + r, ct * 16 + col0, fast_tanh(acc[ct][r]));
}

// ---------------------------------------------------------------------------
// init: wd <- cell_d[:,0,0]; dev_qt <- 0; loss accumulators <- 0; seq out <- 1
// ---------------------------------------------------------------------------
__global__ __launch_bounds__(256) void init_kernel(const float* __restrict__ cell_d,
                                                   float* __restrict__ wd,
                                                   float* __restrict__ dev_qt,
                                                   float* __restrict__ scal,
                                                   float* __restrict__ out_seq) {
  int i = blockIdx.x * blockDim.x + threadIdx.x;
  if (i < N_NODES * EMB) dev_qt[i] = 0.0f;
  if (i < N_NODES) wd[i] = cell_d[(size_t)i * (T_STEPS + 1) * 3];
  if (i < N_NODES * T_STEPS) out_seq[i] = 1.0f;
  if (i < 2) scal[i] = 0.0f;
}

__global__ __launch_bounds__(256) void zero_kernel(float* __restrict__ p, int n) {
  int i = blockIdx.x * blockDim.x + threadIdx.x;
  if (i < n) p[i] = 0.0f;
}

// ---------------------------------------------------------------------------
// rain MLP: cell_out = (tanh(tanh([wd,rain]@W1+b1)@W2+b2))@W3+b3
// ---------------------------------------------------------------------------
#define NWAVES 8
__global__ __launch_bounds__(NWAVES * 32) void rain_kernel(
    const float* __restrict__ cell_d, const float* __restrict__ wd,
    const float* __restrict__ W1, const float* __restrict__ b1,
    const float* __restrict__ W2, const float* __restrict__ b2,
    const float* __restrict__ W3, const float* __restrict__ b3,
    float* __restrict__ cell_out, int t) {
  __shared__ v16h sW2[4 * 32];
  __shared__ v16h sW3[8 * 32];
  __shared__ float sW1[2 * 32], sB1[32], sB2[EMB], sB3[EMB];
  __shared__ v16h stile[NWAVES][2 * 32];

  int tid = threadIdx.x, nthr = blockDim.x;
  stage_wswz32(sW2, W2, tid, nthr);
  stage_wswz(sW3, W3, tid, nthr);
  stage_f32(sW1, W1, 2 * 32, tid, nthr);
  stage_f32(sB1, b1, 32, tid, nthr);
  stage_f32(sB2, b2, EMB, tid, nthr);
  stage_f32(sB3, b3, EMB, tid, nthr);
  __syncthreads();

  int wave = tid >> 5, lane = tid & 31;
  int ntile = blockIdx.x * NWAVES + wave;
  if (ntile >= N_NODES / 16) return;
  int nb = ntile * 16;
  v16h* my = stile[wave];
  _Float16* myh = (_Float16*)my;

  int m = lane & 15;
  int node = nb + m;
  float wdv = wd[node];
  float rnv = cell_d[((size_t)node * (T_STEPS + 1) + t) * 3 + 2];

  // h1 (K=32) built directly in A-fragment layout
  v16h a;
#pragma unroll
  for (int j = 0; j < 16; ++j) {
    int k = koff(lane, j);
    a[j] = (_Float16)fast_tanh(wdv * sW1[k] + rnv * sW1[32 + k] + sB1[k]);
  }
  // h2 = tanh(h1 @ W2 + b2) -> swizzled tile
  {
    int col0 = lane & 15;
    int rbase = (lane & 16) ? 8 : 0;
#pragma unroll
    for (int ct = 0; ct < 4; ++ct) {
      float bv = sB2[ct * 16 + col0];
      v8f c;
#pragma unroll
      for (int r = 0; r < 8; ++r) c[r] = bv;
      c = wmma16(a, sW2[ct * 32 + lane], c);
#pragma unroll
      for (int r = 0; r < 8; ++r)
        put_swz(myh, rbase + r, ct * 16 + col0, fast_tanh(c[r]));
    }
  }
  // rain_emb = h2 @ W3 + b3
  v8f acc[4];
  gemm_frag(my[lane], my[32 + lane], sW3, sB3, lane, acc);

  int col0 = lane & 15;
  int rbase = (lane & 16) ? 8 : 0;
#pragma unroll
  for (int ct = 0; ct < 4; ++ct)
#pragma unroll
    for (int r = 0; r < 8; ++r)
      cell_out[(size_t)(nb + rbase + r) * EMB + ct * 16 + col0] = acc[ct][r];
}

// ---------------------------------------------------------------------------
// edge message kernel: per 16-edge tile
//   slope = x_i - x_j + (dem_i - dem_j)*demW      (dem biases cancel)
//   gate  = sigmoid(tanh(slope@gW1+gb1)@gW2+gb2)
//   msg   = (tanh(x_i@cW1+cb1)@cW2+cb2) * gate
//   dev_qt[e0] += msg  (atomic scatter)
// ---------------------------------------------------------------------------
#define EWAVES 4
__global__ __launch_bounds__(EWAVES * 32) void edge_msg_kernel(
    const float* __restrict__ cell_out, const float* __restrict__ cell_s,
    const int* __restrict__ e0, const int* __restrict__ e1,
    const float* __restrict__ demW,
    const float* __restrict__ gW1, const float* __restrict__ gb1,
    const float* __restrict__ gW2, const float* __restrict__ gb2,
    const float* __restrict__ cW1, const float* __restrict__ cb1,
    const float* __restrict__ cW2, const float* __restrict__ cb2,
    float* __restrict__ dev_qt) {
  __shared__ v16h sGW1[8 * 32], sGW2[8 * 32], sCW1[8 * 32], sCW2[8 * 32];
  __shared__ float sGB1[EMB], sGB2[EMB], sCB1[EMB], sCB2[EMB], sDW[EMB];
  __shared__ v16h tS[EWAVES][2 * 32];   // slope / t1 / t2 (reused)
  __shared__ v16h tXi[EWAVES][2 * 32];  // x_i fragments
  __shared__ int se0[EWAVES][16];

  int tid = threadIdx.x, nthr = blockDim.x;

  // Prefetch the gather rows while we stage weights into LDS.
  {
    int et = blockIdx.x * EWAVES + (tid >> 5);
    if (et * 16 < N_EDGES) {
      int ei = et * 16 + (tid & 15);
      int half = (tid & 16) ? 32 : 0;
      __builtin_prefetch(cell_out + (size_t)e0[ei] * EMB + half, 0, 3);
      __builtin_prefetch(cell_out + (size_t)e1[ei] * EMB + half, 0, 3);
    }
  }

  stage_wswz(sGW1, gW1, tid, nthr);
  stage_wswz(sGW2, gW2, tid, nthr);
  stage_wswz(sCW1, cW1, tid, nthr);
  stage_wswz(sCW2, cW2, tid, nthr);
  stage_f32(sGB1, gb1, EMB, tid, nthr);
  stage_f32(sGB2, gb2, EMB, tid, nthr);
  stage_f32(sCB1, cb1, EMB, tid, nthr);
  stage_f32(sCB2, cb2, EMB, tid, nthr);
  stage_f32(sDW, demW, EMB, tid, nthr);
  __syncthreads();

  int wave = tid >> 5, lane = tid & 31;
  int etile = blockIdx.x * EWAVES + wave;
  if (etile * 16 >= N_EDGES) return;
  int eb = etile * 16;
  _Float16* myS = (_Float16*)tS[wave];
  _Float16* myXi = (_Float16*)tXi[wave];
  int* my_e0 = se0[wave];

  // Stage slope and x_i tiles (lane pair (m, m+16) splits 64 cols of row m),
  // scattering quads directly into swizzled fragment order (b64 stores).
  {
    int m = lane & 15;
    int ei = eb + m;
    int s = e0[ei], d = e1[ei];
    if (lane < 16) my_e0[m] = s;
    float dd = cell_s[(size_t)s * 2] - cell_s[(size_t)d * 2];
    const float* xi = cell_out + (size_t)s * EMB;
    const float* xj = cell_out + (size_t)d * EMB;
    int c0 = (lane & 16) ? 32 : 0;
#pragma unroll
    for (int q = 0; q < 8; ++q) {
      int cc = c0 + q * 4;
      float4 va = *(const float4*)(xi + cc);
      float4 vb = *(const float4*)(xj + cc);
      put_swz4(myXi, m, cc, va);
      float4 vs = {va.x - vb.x + dd * sDW[cc + 0],
                   va.y - vb.y + dd * sDW[cc + 1],
                   va.z - vb.z + dd * sDW[cc + 2],
                   va.w - vb.w + dd * sDW[cc + 3]};
      put_swz4(myS, m, cc, vs);
    }
  }

  v8f gate[4], acc[4];
  // GEMM1: t1 = tanh(slope @ gW1 + gb1) -> back into tS
  gemm_frag(tS[wave][lane], tS[wave][32 + lane], sGW1, sGB1, lane, acc);
  store_swz_tanh(myS, lane, acc);
  // GEMM2: gate = sigmoid(t1 @ gW2 + gb2)   (kept in registers, C layout)
  gemm_frag(tS[wave][lane], tS[wave][32 + lane], sGW2, sGB2, lane, gate);
#pragma unroll
  for (int ct = 0; ct < 4; ++ct)
#pragma unroll
    for (int r = 0; r < 8; ++r) gate[ct][r] = fast_sigmoid(gate[ct][r]);
  // GEMM3: t2 = tanh(x_i @ cW1 + cb1) -> back into tS
  gemm_frag(tXi[wave][lane], tXi[wave][32 + lane], sCW1, sCB1, lane, acc);
  store_swz_tanh(myS, lane, acc);
  // GEMM4: msg = (t2 @ cW2 + cb2) * gate, atomic scatter into dev_qt[e0]
  gemm_frag(tS[wave][lane], tS[wave][32 + lane], sCW2, sCB2, lane, acc);
  {
    int col0 = lane & 15;
    int rbase = (lane & 16) ? 8 : 0;
#pragma unroll
    for (int ct = 0; ct < 4; ++ct)
#pragma unroll
      for (int r = 0; r < 8; ++r) {
        float msg = acc[ct][r] * gate[ct][r];
        int s = my_e0[rbase + r];
        atomicAdd(dev_qt + (size_t)s * EMB + ct * 16 + col0, msg);
      }
  }
}

// ---------------------------------------------------------------------------
// in_qt[e1] += dev_qt[e0]   (segment sum over edges)
// ---------------------------------------------------------------------------
__global__ __launch_bounds__(256) void scatter_inqt_kernel(
    const float* __restrict__ dev_qt, const int* __restrict__ e0,
    const int* __restrict__ e1, float* __restrict__ in_qt) {
  int idx = blockIdx.x * blockDim.x + threadIdx.x;
  if (idx >= N_EDGES * 16) return;
  int e = idx >> 4;
  int q = (idx & 15) << 2;
  int s = e0[e], d = e1[e];
  float4 v = *(const float4*)(dev_qt + (size_t)s * EMB + q);
  float* dst = in_qt + (size_t)d * EMB + q;
  atomicAdd(dst + 0, v.x);
  atomicAdd(dst + 1, v.y);
  atomicAdd(dst + 2, v.z);
  atomicAdd(dst + 3, v.w);
}

// ---------------------------------------------------------------------------
// cell_out += (in_qt - dev_qt) @ dql_W + dql_b ; tanh if not last layer
// ---------------------------------------------------------------------------
__global__ __launch_bounds__(NWAVES * 32) void node_dql_kernel(
    const float* __restrict__ in_qt, const float* __restrict__ dev_qt,
    const float* __restrict__ W, const float* __restrict__ b,
    float* __restrict__ cell_out, int do_tanh) {
  __shared__ v16h sW[8 * 32];
  __shared__ float sB[EMB];
  int tid = threadIdx.x, nthr = blockDim.x;
  stage_wswz(sW, W, tid, nthr);
  stage_f32(sB, b, EMB, tid, nthr);
  __syncthreads();

  int wave = tid >> 5, lane = tid & 31;
  int ntile = blockIdx.x * NWAVES + wave;
  if (ntile >= N_NODES / 16) return;
  int nb = ntile * 16;

  int m = lane & 15;
  size_t row = (size_t)(nb + m) * EMB;
  v16h a0 = afrag_global_diff(in_qt + row, dev_qt + row, lane, 0);
  v16h a1 = afrag_global_diff(in_qt + row, dev_qt + row, lane, 32);
  v8f acc[4];
  gemm_frag(a0, a1, sW, sB, lane, acc);

  int col0 = lane & 15;
  int rbase = (lane & 16) ? 8 : 0;
#pragma unroll
  for (int ct = 0; ct < 4; ++ct)
#pragma unroll
    for (int r = 0; r < 8; ++r) {
      size_t o = (size_t)(nb + rbase + r) * EMB + ct * 16 + col0;
      float v = cell_out[o] + acc[ct][r];
      cell_out[o] = do_tanh ? fast_tanh(v) : v;
    }
}

// ---------------------------------------------------------------------------
// wd += tanh(cell_out @ elev_W1 + b1) @ elev_W2 + b2 ; losses ; out_wd[:,t]
// ---------------------------------------------------------------------------
__global__ __launch_bounds__(NWAVES * 32) void elev_kernel(
    const float* __restrict__ cell_out, const float* __restrict__ cell_d,
    const float* __restrict__ W1, const float* __restrict__ b1,
    const float* __restrict__ W2, const float* __restrict__ b2,
    float* __restrict__ wd, float* __restrict__ out_wd,
    float* __restrict__ scal, int t) {
  __shared__ v16h sW1[8 * 32];
  __shared__ float sB1[EMB], sW2[EMB];
  __shared__ float tile[NWAVES][16 * EMB];  // f32, consumed by scalar dot
  __shared__ float s_acc[2];

  int tid = threadIdx.x, nthr = blockDim.x;
  stage_wswz(sW1, W1, tid, nthr);
  stage_f32(sB1, b1, EMB, tid, nthr);
  stage_f32(sW2, W2, EMB, tid, nthr);
  if (tid < 2) s_acc[tid] = 0.0f;
  __syncthreads();

  int wave = tid >> 5, lane = tid & 31;
  int ntile = blockIdx.x * NWAVES + wave;
  float hub = 0.0f, rel = 0.0f;
  if (ntile < N_NODES / 16) {
    int nb = ntile * 16;
    float* my = tile[wave];

    int m = lane & 15;
    size_t row = (size_t)(nb + m) * EMB;
    v16h a0 = afrag_global(cell_out + row, lane, 0);
    v16h a1 = afrag_global(cell_out + row, lane, 32);
    v8f acc[4];
    gemm_frag(a0, a1, sW1, sB1, lane, acc);
    {
      int col0 = lane & 15;
      int rbase = (lane & 16) ? 8 : 0;
#pragma unroll
      for (int ct = 0; ct < 4; ++ct)
#pragma unroll
        for (int r = 0; r < 8; ++r)
          my[(rbase + r) * EMB + ct * 16 + col0] = fast_tanh(acc[ct][r]);
    }

    if (lane < 16) {
      float s = b2[0];
#pragma unroll
      for (int k = 0; k < EMB; ++k) s += my[m * EMB + k] * sW2[k];
      int node = nb + m;
      float wdn = wd[node] + s;
      wd[node] = wdn;
      out_wd[(size_t)node * T_STEPS + t] = wdn;
      float target = cell_d[((size_t)node * (T_STEPS + 1) + (t + 1)) * 3];
      float diff = target - wdn;
      float ad = fabsf(diff);
      hub = (ad < 1.0f) ? ad : diff * diff;
      rel = fmaxf(-wdn, 0.0f);
    }
  }
  atomicAdd(&s_acc[0], hub);
  atomicAdd(&s_acc[1], rel);
  __syncthreads();
  if (tid == 0) {
    atomicAdd(&scal[0], s_acc[0]);
    atomicAdd(&scal[1], s_acc[1]);
  }
}

__global__ void finalize_kernel(const float* __restrict__ scal,
                                float* __restrict__ out_loss,
                                float* __restrict__ out_nz) {
  out_loss[0] = scal[0] / (float)N_NODES * 0.5f;
  out_nz[0] = scal[1] / (float)N_NODES;
}

// ---------------------------------------------------------------------------
extern "C" void kernel_launch(void* const* d_in, const int* in_sizes, int n_in,
                              void* d_out, int out_size, void* d_ws, size_t ws_size,
                              hipStream_t stream) {
  (void)in_sizes; (void)n_in; (void)out_size; (void)ws_size;
  const float* cell_d = (const float*)d_in[0];
  const float* cell_s = (const float*)d_in[1];
  const int*   eidx   = (const int*)d_in[3];
  const int* e0 = eidx;
  const int* e1 = eidx + N_EDGES;
  const float* rW1 = (const float*)d_in[5];
  const float* rb1 = (const float*)d_in[6];
  const float* rW2 = (const float*)d_in[7];
  const float* rb2 = (const float*)d_in[8];
  const float* rW3 = (const float*)d_in[9];
  const float* rb3 = (const float*)d_in[10];
  const float* eW1 = (const float*)d_in[11];
  const float* eb1 = (const float*)d_in[12];
  const float* eW2 = (const float*)d_in[13];
  const float* eb2 = (const float*)d_in[14];
  const float* demW = (const float*)d_in[15];
  const float* cW1 = (const float*)d_in[17];
  const float* cb1 = (const float*)d_in[18];
  const float* cW2 = (const float*)d_in[19];
  const float* cb2 = (const float*)d_in[20];
  const float* gW1 = (const float*)d_in[21];
  const float* gb1 = (const float*)d_in[22];
  const float* gW2 = (const float*)d_in[23];
  const float* gb2 = (const float*)d_in[24];
  const float* dqW = (const float*)d_in[25];
  const float* dqb = (const float*)d_in[26];

  float* ws = (float*)d_ws;
  float* wd       = ws;
  float* cell_out = wd + N_NODES;
  float* dev_qt   = cell_out + (size_t)N_NODES * EMB;
  float* in_qt    = dev_qt + (size_t)N_NODES * EMB;
  float* scal     = in_qt + (size_t)N_NODES * EMB;

  float* out_wd   = (float*)d_out;
  float* out_loss = out_wd + (size_t)N_NODES * T_STEPS;
  float* out_seq  = out_loss + 1;
  float* out_nz   = out_seq + (size_t)N_NODES * T_STEPS;

  const int NE = N_NODES * EMB;
  init_kernel<<<(NE + 255) / 256, 256, 0, stream>>>(cell_d, wd, dev_qt, scal, out_seq);

  const int ntiles = N_NODES / 16;                          // 1250
  const int nblocks = (ntiles + NWAVES - 1) / NWAVES;       // 157
  const int eblocks = (N_EDGES / 16 + EWAVES - 1) / EWAVES; // 3125
  const int sblocks = (N_EDGES * 16 + 255) / 256;           // 12500

  for (int t = 0; t < T_STEPS; ++t) {
    rain_kernel<<<nblocks, NWAVES * 32, 0, stream>>>(cell_d, wd, rW1, rb1, rW2, rb2,
                                                     rW3, rb3, cell_out, t);
    for (int c = 0; c < NCONS; ++c) {
      const size_t M = (size_t)EMB * EMB;
      edge_msg_kernel<<<eblocks, EWAVES * 32, 0, stream>>>(
          cell_out, cell_s, e0, e1, demW + (size_t)c * EMB,
          gW1 + c * M, gb1 + (size_t)c * EMB, gW2 + c * M, gb2 + (size_t)c * EMB,
          cW1 + c * M, cb1 + (size_t)c * EMB, cW2 + c * M, cb2 + (size_t)c * EMB,
          dev_qt);
      zero_kernel<<<(NE + 255) / 256, 256, 0, stream>>>(in_qt, NE);
      scatter_inqt_kernel<<<sblocks, 256, 0, stream>>>(dev_qt, e0, e1, in_qt);
      node_dql_kernel<<<nblocks, NWAVES * 32, 0, stream>>>(
          in_qt, dev_qt, dqW + c * M, dqb + (size_t)c * EMB, cell_out,
          (c + 1 < NCONS) ? 1 : 0);
    }
    elev_kernel<<<nblocks, NWAVES * 32, 0, stream>>>(cell_out, cell_d, eW1, eb1, eW2,
                                                     eb2, wd, out_wd, scal, t);
  }
  finalize_kernel<<<1, 1, 0, stream>>>(scal, out_loss, out_nz);
}
